// MCC_76252849373756
// MI455X (gfx1250) — matrix operand order repaired and checked by reading, not compiled, not run on hardware
//
#include <hip/hip_runtime.h>
#include <hip/hip_bf16.h>
#include <cstdint>
#include <cstddef>

#define H_  384
#define W_  384
#define HW_ 147456
#define B_  4

typedef __attribute__((ext_vector_type(16))) _Float16 v16h;
typedef __attribute__((ext_vector_type(8)))  float    v8f;

union HU { unsigned int u; _Float16 h[2]; };

__device__ __forceinline__ float gelu_f(float x) {
    return 0.5f * x * (1.0f + erff(x * 0.70710678118654752440f));
}

// ---------------------------------------------------------------- utilities
__global__ void cast_f32_to_f16(const float* __restrict__ src,
                                _Float16* __restrict__ dst, size_t n) {
    size_t i = (size_t)blockIdx.x * blockDim.x + threadIdx.x;
    size_t stride = (size_t)gridDim.x * blockDim.x;
    for (; i < n; i += stride) dst[i] = (_Float16)src[i];
}

__global__ void zero_f32(float* __restrict__ p, int n) {
    int i = blockIdx.x * blockDim.x + threadIdx.x;
    if (i < n) p[i] = 0.f;
}

// ---------------------------------------------------------------- WMMA conv
// Implicit-GEMM conv (KSxKS, pad=KS/2). Block = 256 threads = 8 waves,
// wave tile = 16 couts x 16 pixels. CT cout-tiles x PT pixel-tiles = 8.
// EPI: 0=store f16, 1=bias+relu f16, 2=gelu f16, 3=+resid(f32) f16,
//      4=bias+relu+global-pool accumulate (no store).
// SEL: pick (wA,biasA) vs (wB,biasB) from *pred (==2 -> B).
// KS==1 path fills the LDS tile with GLOBAL_LOAD_ASYNC_TO_LDS_B128
// (contiguous rows, no halo -> no zero-fill needed), drained on ASYNCcnt.
template<int CIN, int KS, int CT, int PT, int EPI, bool SEL>
__global__ __launch_bounds__(256) void conv_wmma(
    const _Float16* __restrict__ in, const _Float16* __restrict__ wA,
    const _Float16* __restrict__ wB, const float* __restrict__ biasA,
    const float* __restrict__ biasB, const int* __restrict__ pred,
    _Float16* __restrict__ out, const float* __restrict__ resid,
    float* __restrict__ pooled, int Cout, int iters)
{
    constexpr int SW  = PT * 16;
    constexpr int TW  = SW + KS - 1;
    constexpr int PAD = KS / 2;
    constexpr int K   = CIN * KS * KS;
    constexpr int CH  = K / 32;
    constexpr int SPR = W_ / SW;
    constexpr int NSTRIP = SPR * H_;

    __shared__ alignas(16) _Float16 tile[CIN * KS * TW];
    __shared__ float pacc[CT * 16];

    const int tid  = threadIdx.x;
    const int wave = tid >> 5;
    const int lane = tid & 31;
    const int hi   = lane >> 4;
    const int lm   = lane & 15;
    const int ct   = wave % CT;
    const int pt   = wave / CT;
    const int b    = blockIdx.z;
    const int coutBlk = blockIdx.y * (CT * 16);

    const _Float16* w = wA;
    const float* bias = biasA;
    if (SEL && *pred == 2) { w = wB; bias = biasB; }

    if (EPI == 4) {
        for (int i = tid; i < CT * 16; i += 256) pacc[i] = 0.f;
    }
    __syncthreads();

    const int coutRowA = coutBlk + ct * 16 + lm;   // A fragment: lane holds M=lm
    const unsigned int* wrow = (const unsigned int*)(w + (size_t)coutRowA * K);
    const int nloc = pt * 16 + lm;                 // B/D fragment: N = lane&15

    for (int it = 0; it < iters; ++it) {
        int s = blockIdx.x * iters + it;
        if (s >= NSTRIP) break;
        int sy = s / SPR;
        int sx = (s % SPR) * SW;

        if (KS == 1 && (CIN * TW) == 2048) {
            // Async global->LDS fill: 256 lanes x 16B, rows are contiguous.
            int ci  = tid >> 3;
            int ch8 = tid & 7;
            unsigned int ldsoff =
                (unsigned int)(uintptr_t)(&tile[0]) + (unsigned int)(ci * TW * 2 + ch8 * 16);
            unsigned int goff =
                (unsigned int)((((size_t)b * CIN + ci) * HW_ + (size_t)sy * W_ + sx) * 2
                               + (size_t)(ch8 * 16));
            asm volatile("global_load_async_to_lds_b128 %0, %1, %2 offset:0"
                         :: "v"(ldsoff), "v"(goff), "s"(in) : "memory");
            asm volatile("s_wait_asynccnt 0x0" ::: "memory");
        } else {
            // cooperative halo tile load: [CIN][KS rows][TW cols]
            for (int i = tid; i < CIN * KS * TW; i += 256) {
                int ci = i / (KS * TW);
                int r  = i - ci * (KS * TW);
                int ky = r / TW;
                int dx = r - ky * TW;
                int y = sy + ky - PAD;
                int x = sx + dx - PAD;
                _Float16 val = (_Float16)0.f;
                if ((unsigned)y < (unsigned)H_ && (unsigned)x < (unsigned)W_)
                    val = in[(((size_t)b * CIN + ci) * H_ + y) * W_ + x];
                tile[i] = val;
            }
        }
        __syncthreads();

        v8f acc = {0.f, 0.f, 0.f, 0.f, 0.f, 0.f, 0.f, 0.f};
        #pragma unroll 3
        for (int kk = 0; kk < CH; ++kk) {
            v16h af, bf;
            // A: 16x32 f16, lane<16: K 0..15; lane>=16: K 16..31
            #pragma unroll
            for (int v = 0; v < 8; ++v) {
                int k0 = (hi << 4) + (v << 1);
                HU hu; hu.u = wrow[(kk * 32 + k0) >> 1];
                af[2 * v]     = hu.h[0];
                af[2 * v + 1] = hu.h[1];
            }
            // B: 32x16 f16, lane: N=lm, K = hi*16 + 2v + e (im2col from LDS)
            int kb = kk * 32 + hi * 16;
            #pragma unroll
            for (int v = 0; v < 8; ++v) {
                #pragma unroll
                for (int e = 0; e < 2; ++e) {
                    int gk = kb + 2 * v + e;
                    int ci = gk / (KS * KS);
                    int j  = gk - ci * (KS * KS);
                    int ky = j / KS;
                    int kx = j - ky * KS;
                    bf[2 * v + e] = tile[(ci * KS + ky) * TW + nloc + kx];
                }
            }
            acc = __builtin_amdgcn_wmma_f32_16x16x32_f16(
                false, af, false, bf, (short)0, acc, false, false);
        }
        __syncthreads();   // tile consumed; safe to refill next iteration

        // epilogue: D layout M = v + 8*hi, N = lm
        #pragma unroll
        for (int v = 0; v < 8; ++v) {
            int m  = v + hi * 8;
            int cg = coutBlk + ct * 16 + m;
            float val = acc[v];
            if (EPI == 1 || EPI == 4) {
                if (bias) val += bias[cg];
                val = fmaxf(val, 0.f);
            }
            if (EPI == 2) val = gelu_f(val);
            size_t oi = (((size_t)b * Cout + cg) * H_ + sy) * W_ + sx + nloc;
            if (EPI == 3) val += resid[oi];
            if (EPI == 4) {
                float t = val;       // reduce over 16 pixels (lanes in half-group)
                #pragma unroll
                for (int sft = 8; sft >= 1; sft >>= 1) t += __shfl_xor(t, sft, 32);
                if (lm == 0) atomicAdd(&pacc[ct * 16 + m], t);
            } else {
                out[oi] = (_Float16)val;
            }
        }
    }

    if (EPI == 4) {
        __syncthreads();
        for (int i = tid; i < CT * 16; i += 256)
            atomicAdd(&pooled[(size_t)b * Cout + coutBlk + i], pacc[i]);
    }
}

// ---------------------------------------------------------------- depthwise
template<bool GELU, bool OUTF32>
__global__ __launch_bounds__(256) void dwconv3x3(
    const _Float16* __restrict__ in, const float* __restrict__ w,
    void* __restrict__ outv, int C)
{
    size_t t = (size_t)blockIdx.x * 256 + threadIdx.x;   // over B*C*H*W (exact)
    int x = (int)(t % W_);
    size_t q = t / W_;
    int y = (int)(q % H_);
    size_t bc = q / H_;
    int c = (int)(bc % C);
    const _Float16* ib = in + bc * (size_t)HW_;
    const float* wc = w + c * 9;
    float acc = 0.f;
    #pragma unroll
    for (int ky = 0; ky < 3; ++ky) {
        int yy = y + ky - 1;
        if ((unsigned)yy >= (unsigned)H_) continue;
        #pragma unroll
        for (int kx = 0; kx < 3; ++kx) {
            int xx = x + kx - 1;
            if ((unsigned)xx >= (unsigned)W_) continue;
            acc += wc[ky * 3 + kx] * (float)ib[yy * W_ + xx];
        }
    }
    if (GELU) acc = gelu_f(acc);
    if (OUTF32) ((float*)outv)[t] = acc;
    else        ((_Float16*)outv)[t] = (_Float16)acc;
}

// ---------------------------------------------------------------- detector FC
__global__ void fc_argmax(const float* __restrict__ pooled,
                          const float* __restrict__ fw,
                          const float* __restrict__ fb, int* __restrict__ pred)
{
    __shared__ float logits[3];
    int t = threadIdx.x;
    if (t < 3) {
        float s = fb[t];
        const float inv = 1.f / (float)HW_;
        for (int i = 0; i < 256; ++i) s += pooled[i] * inv * fw[t * 256 + i];
        logits[t] = s;
    }
    __syncthreads();
    if (t == 0) {
        int p = 0;
        if (logits[1] > logits[0]) p = 1;
        if (logits[2] > logits[p]) p = 2;
        *pred = p;
    }
}

// ---------------------------------------------------------------- branch sel
__global__ __launch_bounds__(256) void select_x(
    const float* __restrict__ x, const _Float16* __restrict__ br,
    const int* __restrict__ pred, float* __restrict__ xs)
{
    size_t i = (size_t)blockIdx.x * 256 + threadIdx.x;   // B*32*HW exact
    xs[i] = (*pred == 0) ? x[i] : (float)br[i];
}

// ---------------------------------------------------------------- LayerNorm
__global__ __launch_bounds__(256) void layernorm_cf(
    const float* __restrict__ xs, const float* __restrict__ g,
    const float* __restrict__ be, _Float16* __restrict__ a16)
{
    size_t idx = (size_t)blockIdx.x * 256 + threadIdx.x;  // over B*HW
    size_t b = idx / HW_;
    size_t n = idx - b * HW_;
    const float* p = xs + b * 32 * (size_t)HW_ + n;
    float mean = 0.f;
    #pragma unroll
    for (int c = 0; c < 32; ++c) mean += p[(size_t)c * HW_];
    mean *= (1.f / 32.f);
    float var = 0.f;
    #pragma unroll
    for (int c = 0; c < 32; ++c) { float d = p[(size_t)c * HW_] - mean; var += d * d; }
    var *= (1.f / 32.f);
    float rstd = rsqrtf(var + 1e-6f);
    _Float16* o = a16 + b * 32 * (size_t)HW_ + n;
    #pragma unroll
    for (int c = 0; c < 32; ++c)
        o[(size_t)c * HW_] = (_Float16)(g[c] * ((p[(size_t)c * HW_] - mean) * rstd) + be[c]);
}

// ------------------------------------------------- channel attention reduce
// grid (B*HEADS, SPLIT). Per (b,h): accumulate q.k Gram (4x4) + |q|^2,|k|^2.
__global__ __launch_bounds__(256) void attn_reduce(
    const _Float16* __restrict__ qkv, float* __restrict__ raw,
    float* __restrict__ qss, float* __restrict__ kss)
{
    __shared__ float red[256];
    int bh = blockIdx.x;
    int b = bh >> 3;
    int h = bh & 7;
    const _Float16* qb = qkv + ((size_t)b * 96 + h * 4) * HW_;
    const _Float16* kb = qkv + ((size_t)b * 96 + 32 + h * 4) * HW_;
    float aqk[16], aq[4], ak[4];
    #pragma unroll
    for (int i = 0; i < 16; ++i) aqk[i] = 0.f;
    #pragma unroll
    for (int i = 0; i < 4; ++i) { aq[i] = 0.f; ak[i] = 0.f; }
    for (int n = blockIdx.y * 256 + threadIdx.x; n < HW_; n += gridDim.y * 256) {
        float qv[4], kv[4];
        #pragma unroll
        for (int c = 0; c < 4; ++c) {
            qv[c] = (float)qb[(size_t)c * HW_ + n];
            kv[c] = (float)kb[(size_t)c * HW_ + n];
        }
        #pragma unroll
        for (int c = 0; c < 4; ++c) {
            aq[c] += qv[c] * qv[c];
            ak[c] += kv[c] * kv[c];
            #pragma unroll
            for (int d = 0; d < 4; ++d) aqk[c * 4 + d] += qv[c] * kv[d];
        }
    }
    #pragma unroll
    for (int v = 0; v < 24; ++v) {
        float val = (v < 16) ? aqk[v] : ((v < 20) ? aq[v - 16] : ak[v - 20]);
        red[threadIdx.x] = val;
        __syncthreads();
        for (int s = 128; s > 0; s >>= 1) {
            if (threadIdx.x < s) red[threadIdx.x] += red[threadIdx.x + s];
            __syncthreads();
        }
        if (threadIdx.x == 0) {
            if (v < 16)      atomicAdd(&raw[bh * 16 + v], red[0]);
            else if (v < 20) atomicAdd(&qss[bh * 4 + (v - 16)], red[0]);
            else             atomicAdd(&kss[bh * 4 + (v - 20)], red[0]);
        }
        __syncthreads();
    }
}

__global__ void attn_softmax(const float* __restrict__ raw,
                             const float* __restrict__ qss,
                             const float* __restrict__ kss,
                             const float* __restrict__ temp,
                             float* __restrict__ attn)
{
    int i = threadIdx.x;           // 128 = B*8*4 (b,h,c)
    if (i >= 128) return;
    int c = i & 3; int bh = i >> 2; int h = bh & 7;
    float qn = fmaxf(sqrtf(qss[bh * 4 + c]), 1e-12f);
    float t = temp[h];
    float v[4]; float mx = -1e30f;
    #pragma unroll
    for (int d = 0; d < 4; ++d) {
        float kn = fmaxf(sqrtf(kss[bh * 4 + d]), 1e-12f);
        v[d] = raw[bh * 16 + c * 4 + d] / (qn * kn) * t;
        mx = fmaxf(mx, v[d]);
    }
    float s = 0.f;
    #pragma unroll
    for (int d = 0; d < 4; ++d) { v[d] = expf(v[d] - mx); s += v[d]; }
    float inv = 1.f / s;
    #pragma unroll
    for (int d = 0; d < 4; ++d) attn[bh * 16 + c * 4 + d] = v[d] * inv;
}

__global__ __launch_bounds__(256) void attn_apply(
    const _Float16* __restrict__ qkv, const float* __restrict__ attn,
    _Float16* __restrict__ out)
{
    size_t i = (size_t)blockIdx.x * 256 + threadIdx.x;   // B*32*HW exact
    size_t n = i % HW_;
    size_t bc = i / HW_;
    int cg = (int)(bc % 32);
    int b  = (int)(bc / 32);
    int h = cg >> 2; int c = cg & 3;
    const _Float16* vb = qkv + ((size_t)b * 96 + 64 + h * 4) * HW_ + n;
    const float* a = attn + ((b * 8 + h) * 4 + c) * 4;
    float s = a[0] * (float)vb[0]
            + a[1] * (float)vb[(size_t)HW_]
            + a[2] * (float)vb[2 * (size_t)HW_]
            + a[3] * (float)vb[3 * (size_t)HW_];
    out[i] = (_Float16)s;
}

// ---------------------------------------------------------------- launcher
extern "C" void kernel_launch(void* const* d_in, const int* in_sizes, int n_in,
                              void* d_out, int out_size, void* d_ws, size_t ws_size,
                              hipStream_t stream)
{
    const float* x        = (const float*)d_in[0];
    const float* ln_w     = (const float*)d_in[1];
    const float* ln_b     = (const float*)d_in[2];
    const float* temper   = (const float*)d_in[3];
    const float* pw_w     = (const float*)d_in[4];
    const float* dw_w     = (const float*)d_in[5];
    const float* proj_w   = (const float*)d_in[6];
    const float* ff1_w    = (const float*)d_in[7];
    const float* ffdw_w   = (const float*)d_in[8];
    const float* det_w1   = (const float*)d_in[9];
    const float* det_b1   = (const float*)d_in[10];
    const float* det_w2   = (const float*)d_in[11];
    const float* det_b2   = (const float*)d_in[12];
    const float* det_w3   = (const float*)d_in[13];
    const float* det_b3   = (const float*)d_in[14];
    const float* det_fc_w = (const float*)d_in[15];
    const float* det_fc_b = (const float*)d_in[16];
    const float* hvi_w1   = (const float*)d_in[17];
    const float* hvi_b1   = (const float*)d_in[18];
    const float* hvi_w2   = (const float*)d_in[19];
    const float* hvi_b2   = (const float*)d_in[20];
    const float* ycc_w1   = (const float*)d_in[21];
    const float* ycc_b1   = (const float*)d_in[22];
    const float* ycc_w2   = (const float*)d_in[23];
    const float* ycc_b2   = (const float*)d_in[24];

    const size_t BHW = (size_t)B_ * HW_;
    char* ws = (char*)d_ws;

    // --- f16 weight pool (element offsets) ---
    _Float16* w16 = (_Float16*)ws;
    const size_t o_det1 = 0,      o_det2 = 18432,  o_det3 = 92160,
                 o_hvi1 = 387072, o_ycc1 = 405504, o_hvi2 = 423936,
                 o_ycc2 = 442368, o_pw   = 460800, o_proj = 463872,
                 o_ff1  = 464896;
    const size_t W16_BYTES = (size_t)1 << 20;

    // --- small scratch (64 KB, zeroed every call) ---
    char* sc = ws + W16_BYTES;
    float* pooled   = (float*)sc;                 // [B][256]
    int*   pred     = (int*)(sc + 4096);
    float* attn_raw = (float*)(sc + 4352);        // 512
    float* qss      = (float*)(sc + 6400);        // 128
    float* kss      = (float*)(sc + 6912);        // 128
    float* attnW    = (float*)(sc + 7424);        // 512

    // --- big regions (reused across stages) ---
    char* big = ws + W16_BYTES + 65536;
    _Float16* RX16 = (_Float16*)big;                                   // 32ch f16: x16 -> a16 -> attn_out16
    _Float16* R1   = (_Float16*)(big + BHW * 32 * 2);                  // 64ch f16: f1 -> branch1 -> y16
    _Float16* R2   = (_Float16*)(big + BHW * 32 * 2 + BHW * 64 * 2);   // 128ch f16: f2 -> qkv_pre -> ff1_out
    _Float16* R3   = (_Float16*)(big + BHW * 32 * 2 + BHW * 64 * 2 + BHW * 128 * 2); // 96ch f16
    float*    XSEL = (float*)(big + BHW * 32 * 2 + BHW * 64 * 2 + BHW * 128 * 2 + BHW * 96 * 2); // 32ch f32

    // 0) zero small scratch (pooled/attn accumulators)
    zero_f32<<<64, 256, 0, stream>>>((float*)sc, 16384);

    // 1) casts to f16
    auto cast = [&](const float* s, _Float16* dptr, size_t n) {
        int blocks = (int)((n + 255) / 256);
        if (blocks > 8192) blocks = 8192;
        cast_f32_to_f16<<<blocks, 256, 0, stream>>>(s, dptr, n);
    };
    cast(x, RX16, BHW * 32);
    cast(det_w1, w16 + o_det1, 18432);
    cast(det_w2, w16 + o_det2, 73728);
    cast(det_w3, w16 + o_det3, 294912);
    cast(hvi_w1, w16 + o_hvi1, 18432);
    cast(ycc_w1, w16 + o_ycc1, 18432);
    cast(hvi_w2, w16 + o_hvi2, 18432);
    cast(ycc_w2, w16 + o_ycc2, 18432);
    cast(pw_w,   w16 + o_pw,   3072);
    cast(proj_w, w16 + o_proj, 1024);
    cast(ff1_w,  w16 + o_ff1,  1024);

    // 2) detector (batch 0 only — reference uses logits[0] only)
    conv_wmma<32, 3, 4, 2, 1, false><<<dim3(4608, 1, 1), 256, 0, stream>>>(
        RX16, w16 + o_det1, nullptr, det_b1, nullptr, nullptr, R1, nullptr, nullptr, 64, 1);
    conv_wmma<64, 3, 4, 2, 1, false><<<dim3(4608, 2, 1), 256, 0, stream>>>(
        R1, w16 + o_det2, nullptr, det_b2, nullptr, nullptr, R2, nullptr, nullptr, 128, 1);
    // conv3 fused with ReLU + global pooling (never materializes f3)
    conv_wmma<128, 3, 4, 2, 4, false><<<dim3(288, 4, 1), 256, 0, stream>>>(
        R2, w16 + o_det3, nullptr, det_b3, nullptr, nullptr, nullptr, nullptr, pooled, 256, 16);
    fc_argmax<<<1, 256, 0, stream>>>(pooled, det_fc_w, det_fc_b, pred);

    // 3) color-space branch (device-side weight selection by pred)
    conv_wmma<32, 3, 4, 2, 1, true><<<dim3(4608, 1, B_), 256, 0, stream>>>(
        RX16, w16 + o_hvi1, w16 + o_ycc1, hvi_b1, ycc_b1, pred, R1, nullptr, nullptr, 64, 1);
    conv_wmma<64, 3, 2, 4, 1, true><<<dim3(2304, 1, B_), 256, 0, stream>>>(
        R1, w16 + o_hvi2, w16 + o_ycc2, hvi_b2, ycc_b2, pred, R3, nullptr, nullptr, 32, 1);
    select_x<<<73728, 256, 0, stream>>>(x, R3, pred, XSEL);

    // 4) channels-first LayerNorm -> a16
    layernorm_cf<<<2304, 256, 0, stream>>>(XSEL, ln_w, ln_b, RX16);

    // 5) qkv: 1x1 pwconv (WMMA, async-LDS fill) then 3x3 depthwise
    conv_wmma<32, 1, 2, 4, 0, false><<<dim3(2304, 3, B_), 256, 0, stream>>>(
        RX16, w16 + o_pw, nullptr, nullptr, nullptr, nullptr, R2, nullptr, nullptr, 96, 1);
    dwconv3x3<false, false><<<221184, 256, 0, stream>>>(R2, dw_w, (void*)R3, 96);

    // 6) channel-wise transposed attention (4x4 per head)
    attn_reduce<<<dim3(32, 32), 256, 0, stream>>>(R3, attn_raw, qss, kss);
    attn_softmax<<<1, 128, 0, stream>>>(attn_raw, qss, kss, temper, attnW);
    attn_apply<<<73728, 256, 0, stream>>>(R3, attnW, RX16);

    // 7) proj 1x1 (WMMA, async-LDS fill) fused with residual add -> y16
    conv_wmma<32, 1, 2, 4, 3, false><<<dim3(2304, 1, B_), 256, 0, stream>>>(
        RX16, w16 + o_proj, nullptr, nullptr, nullptr, nullptr, R1, XSEL, nullptr, 32, 1);

    // 8) feedforward: 1x1 (WMMA, async-LDS fill) + exact GELU, then depthwise 3x3 + GELU -> f32 out
    conv_wmma<32, 1, 2, 4, 2, false><<<dim3(2304, 1, B_), 256, 0, stream>>>(
        R1, w16 + o_ff1, nullptr, nullptr, nullptr, nullptr, R2, nullptr, nullptr, 32, 1);
    dwconv3x3<true, true><<<73728, 256, 0, stream>>>(R2, ffdw_w, d_out, 32);
}
